// ImplicitFlowDecoder_82712480187035
// MI455X (gfx1250) — matrix operand order repaired and checked by reading, not compiled, not run on hardware
//
#include <hip/hip_runtime.h>
#include <cstdint>
#include <cstddef>

typedef __attribute__((ext_vector_type(16))) _Float16 v16h;
typedef __attribute__((ext_vector_type(8)))  float    v8f;

#define EPSF 1e-6f
#define HF 384
#define WF 512
#define NPB (HF*WF)   // 196608 points per batch
#define SBA 288       // bufA stride (halves) -- holds 288-wide mlp_in
#define SBB 256       // bufB stride (halves)
#define SG  32        // geom stride (floats per point)

union HV { v16h v; uint4 q[2]; };
union PK2 { _Float16 h[2]; unsigned u; };

// ---------------------------------------------------------------------------
// helpers
// ---------------------------------------------------------------------------
__device__ __forceinline__ float unnorm_coord(float cn, float S) {
  float x = ((cn + 1.f) * S - 1.f) * 0.5f;
  return fminf(fmaxf(x, 0.f), S - 1.f);
}

// precompute bilinear quad: 4 plane-local indices (bit-cast) + 4 weights
__device__ __forceinline__ void make_quad(float x, float y, int Wd, int Hd, float* g) {
  float xf = floorf(x), yf = floorf(y);
  int ix0 = (int)xf, iy0 = (int)yf;
  int ix1 = ix0 + 1; if (ix1 > Wd - 1) ix1 = Wd - 1;
  int iy1 = iy0 + 1; if (iy1 > Hd - 1) iy1 = Hd - 1;
  float wx = x - xf, wy = y - yf;
  int r0 = iy0 * Wd, r1 = iy1 * Wd;
  g[0] = __int_as_float(r0 + ix0);
  g[1] = __int_as_float(r0 + ix1);
  g[2] = __int_as_float(r1 + ix0);
  g[3] = __int_as_float(r1 + ix1);
  g[4] = (1.f - wx) * (1.f - wy);
  g[5] = wx * (1.f - wy);
  g[6] = (1.f - wx) * wy;
  g[7] = wx * wy;
}

__device__ __forceinline__ float quad_sample(const float* __restrict__ fc,
                                             int i00, int i01, int i10, int i11,
                                             float w00, float w01, float w10, float w11) {
  return fc[i00]*w00 + fc[i01]*w01 + fc[i10]*w10 + fc[i11]*w11;
}

// cooperative bilinear gather of a 16-point x C tile into LDS (f16).
// Each lane owns one point (pt = lane&15) and a fixed channel phase; the quad
// is loaded once per lane (two ds_load_b128) and reused for all channels.
// Two consecutive channels are packed per ds_store_b32.
__device__ __forceinline__ void gather16(const float* __restrict__ feat, int C, int HW,
                                         const float* geom, int qoff,
                                         _Float16* dst, int dstStride, int lane) {
  const int pt = lane & 15;
  const int cb = (lane >> 4) << 1;          // 0 or 2
  const float* g = geom + pt * SG + qoff;   // 16B aligned
  uint4  iq = *(const uint4*)g;
  float4 wq = *(const float4*)(g + 4);
  const int i00 = (int)iq.x, i01 = (int)iq.y, i10 = (int)iq.z, i11 = (int)iq.w;
  _Float16* drow = dst + pt * dstStride;
  for (int c = cb; c < C; c += 4) {
    const float* fc = feat + (size_t)c * HW;
    float v0 = quad_sample(fc,      i00, i01, i10, i11, wq.x, wq.y, wq.z, wq.w);
    float v1 = quad_sample(fc + HW, i00, i01, i10, i11, wq.x, wq.y, wq.z, wq.w);
    PK2 p; p.h[0] = (_Float16)v0; p.h[1] = (_Float16)v1;
    *(unsigned*)(drow + c) = p.u;           // 4B aligned (c even, stride even)
  }
}

// ---------------------------------------------------------------------------
// wave-level WMMA GEMM: [16 x K] (LDS, stride sa) @ [K x Nout] (packed global)
// ACT: 0 = none, 1 = exact gelu, 2 = relu. addbuf: optional 16x128 residual.
// ---------------------------------------------------------------------------
template<int ACT>
__device__ __forceinline__ void wave_gemm(const _Float16* A, int sa, int K,
                                          const _Float16* __restrict__ Bw,
                                          const float* __restrict__ bias,
                                          _Float16* out, int so, int Nout,
                                          const _Float16* addbuf, int lane) {
  const int m     = lane & 15;
  const int koff  = (lane >> 4) << 3;   // 0 or 8 (A-layout half-lane K shift)
  const int rbase = koff;               // C-layout row base: 0 or 8
  const int NT = Nout >> 4;
  const int KT = K >> 5;
  __builtin_prefetch(Bw + (size_t)lane * 16, 0, 1);
  for (int nt = 0; nt < NT; ++nt) {
    v8f acc = {0.f,0.f,0.f,0.f,0.f,0.f,0.f,0.f};
    for (int kt = 0; kt < KT; ++kt) {
      HV a, b;
      const _Float16* ap = A + m*sa + kt*32 + koff;       // 16B aligned
      a.q[0] = *(const uint4*)ap;                         // K halves 0..7 (+koff)
      a.q[1] = *(const uint4*)(ap + 16);                  // K halves 16..23 (+koff)
      const uint4* bp = (const uint4*)(Bw + ((size_t)((nt*KT + kt)*32 + lane)) * 16);
      b.q[0] = bp[0]; b.q[1] = bp[1];
      acc = __builtin_amdgcn_wmma_f32_16x16x32_f16(false, a.v, false, b.v,
                                                   (short)0, acc, false, false);
    }
    const int col = nt*16 + m;
    const float bvv = bias[col];
    float addv[8];
    if (addbuf) {                         // hoist residual loads: one wait, not 8
#pragma unroll
      for (int r = 0; r < 8; ++r) addv[r] = (float)addbuf[(rbase + r)*128 + col];
    }
#pragma unroll
    for (int r = 0; r < 8; ++r) {
      float v = acc[r] + bvv;
      if (addbuf) v += addv[r];
      if (ACT == 1) v = 0.5f*v*(1.f + erff(v*0.70710678118f));
      if (ACT == 2) v = fmaxf(v, 0.f);
      out[(rbase + r)*so + col] = (_Float16)v;
    }
  }
}

// ---------------------------------------------------------------------------
// weight/bias packing (f32 -> f16, WMMA-B lane layout, gate folding)
// packed element idx = ((nt*KT + kt)*32 + lane)*16 + h
// lane<16 : B[kt*32 + (h<8 ? h : 8+h)][nt*16 + lane]
// lane>=16: B[kt*32 + (h<8 ? 8+h : 16+h)][nt*16 + lane-16]
// ---------------------------------------------------------------------------
__global__ void pack_w_kernel(const float* __restrict__ W, const float* __restrict__ gate,
                              int Kr, int Nr, int Kp, int Np, _Float16* __restrict__ dst) {
  int KT = Kp >> 5, NT = Np >> 4;
  int total = KT * NT * 512;
  for (int idx = blockIdx.x*blockDim.x + threadIdx.x; idx < total;
       idx += gridDim.x*blockDim.x) {
    int t = idx >> 9, rem = idx & 511;
    int lane = rem >> 4, h = rem & 15;
    int kt = t % KT, nt = t / KT;
    int n = nt*16 + (lane & 15);
    int k = kt*32 + ((lane >> 4) << 3) + (h & 7) + ((h >> 3) << 4);
    float v = 0.f;
    if (k < Kr && n < Nr) {
      v = W[(size_t)k * Nr + n];
      if (gate) v *= 1.f / (1.f + expf(-gate[n]));
    }
    dst[idx] = (_Float16)v;
  }
}

__global__ void pack_b_kernel(const float* __restrict__ b, const float* __restrict__ gate,
                              int Nr, int Np, float* __restrict__ dst) {
  int idx = blockIdx.x*blockDim.x + threadIdx.x;
  if (idx < Np) {
    float v = (idx < Nr) ? b[idx] : 0.f;
    if (gate && idx < Nr) v *= 1.f / (1.f + expf(-gate[idx]));
    dst[idx] = v;
  }
}

// ---------------------------------------------------------------------------
// fused decoder kernel: 2 waves/block, 16 points per wave
// ---------------------------------------------------------------------------
__global__ __launch_bounds__(64) void decoder_kernel(
    const float* __restrict__ feat_s8, const float* __restrict__ feat1_s8,
    const float* __restrict__ feat_s16, const float* __restrict__ ctx_s8,
    const float* __restrict__ coarse_flow,
    const _Float16* __restrict__ Wws, const float* __restrict__ Bws,
    float* __restrict__ out) {
  struct __align__(16) WaveScratch {
    _Float16 bufA[16*SBA];      // 9216 B
    _Float16 bufB[16*SBB];      // 8192 B
    _Float16 f8b [16*128];      // 4096 B
    _Float16 f16b[16*128];      // 4096 B
    _Float16 f1wb[16*128];      // 4096 B
    float    geom[16*SG];       // 2048 B
  };
  __shared__ WaveScratch sw[2]; // 63488 B total
  const int lane = threadIdx.x & 31;
  const int wave = threadIdx.x >> 5;
  WaveScratch* s = &sw[wave];
  const int pt0  = (blockIdx.x * 2 + wave) * 16;
  const int bat  = pt0 / NPB;              // tiles never straddle batches
  const int rem0 = pt0 - bat * NPB;

  // ---- phase 0: per-point geometry + gather-quad precompute (lanes 0..15) ----
  if (lane < 16) {
    int rem = rem0 + lane;
    int gy = rem / WF, gx = rem - gy*WF;
    float cnx = fminf(fmaxf(2.f*((float)gx + 0.5f)/(float)WF - 1.f, -1.f+EPSF), 1.f-EPSF);
    float cny = fminf(fmaxf(2.f*((float)gy + 0.5f)/(float)HF - 1.f, -1.f+EPSF), 1.f-EPSF);
    float* g = s->geom + lane*SG;
    // s8 quad (48x64) at slots 0..7 ; s16 quad (24x32) at 8..15
    float x8  = unnorm_coord(cnx, 64.f), y8  = unnorm_coord(cny, 48.f);
    float x16 = unnorm_coord(cnx, 32.f), y16 = unnorm_coord(cny, 24.f);
    make_quad(x8,  y8,  64, 48, g + 0);
    make_quad(x16, y16, 32, 24, g + 8);
    // coarse flow sample reuses the s8 quad
    const float* cf = coarse_flow + (size_t)bat * 2 * 48 * 64;
    int i00 = __float_as_int(g[0]), i01 = __float_as_int(g[1]);
    int i10 = __float_as_int(g[2]), i11 = __float_as_int(g[3]);
    float cpx = quad_sample(cf,         i00,i01,i10,i11, g[4],g[5],g[6],g[7]) * 8.f;
    float cpy = quad_sample(cf + 48*64, i00,i01,i10,i11, g[4],g[5],g[6],g[7]) * 8.f;
    // warped s8 quad at slots 16..23
    float wcx = fminf(fmaxf(cnx + 2.f*cpx/(float)WF, -1.f+EPSF), 1.f-EPSF);
    float wcy = fminf(fmaxf(cny + 2.f*cpy/(float)HF, -1.f+EPSF), 1.f-EPSF);
    make_quad(unnorm_coord(wcx, 64.f), unnorm_coord(wcy, 48.f), 64, 48, g + 16);
    g[24] = cnx; g[25] = cny;
    g[26] = cpx/(float)WF; g[27] = cpy/(float)HF;
    g[28] = cpx; g[29] = cpy;
  }
  __syncthreads();

  // ---- phase 1: cooperative bilinear gathers into LDS (f16) ----
  gather16(ctx_s8  + (size_t)bat* 64*48*64,  64, 48*64, s->geom,  0, s->bufA, SBA, lane);
  gather16(feat_s8 + (size_t)bat*128*48*64, 128, 48*64, s->geom,  0, s->f8b,  128, lane);
  gather16(feat_s16+ (size_t)bat*128*24*32, 128, 24*32, s->geom,  8, s->f16b, 128, lane);
  gather16(feat1_s8+ (size_t)bat*128*48*64, 128, 48*64, s->geom, 16, s->f1wb, 128, lane);
  __syncthreads();

  // ---- phase 2: WMMA GEMM chain (ping-pong bufA <-> bufB) ----
  wave_gemm<0>(s->bufA, SBA,  64, Wws+0,      Bws+0,    s->bufB, SBB, 128, s->f8b,  lane);
  __syncthreads();
  wave_gemm<1>(s->bufB, SBB, 128, Wws+8192,   Bws+128,  s->bufA, SBA, 256, nullptr, lane);
  __syncthreads();
  wave_gemm<0>(s->bufA, SBA, 256, Wws+40960,  Bws+384,  s->bufB, SBB, 128, nullptr, lane);
  __syncthreads();
  wave_gemm<0>(s->bufB, SBB, 128, Wws+73728,  Bws+512,  s->bufA, SBA, 128, s->f16b, lane);
  __syncthreads();
  wave_gemm<1>(s->bufA, SBA, 128, Wws+90112,  Bws+640,  s->bufB, SBB, 256, nullptr, lane);
  __syncthreads();
  wave_gemm<0>(s->bufB, SBB, 256, Wws+122880, Bws+896,  s->bufA, SBA, 128, nullptr, lane);
  __syncthreads();

  // ---- phase 3: concat [fused | f1w | cn | coarse_norm | zeros] into bufA ----
  for (int i = lane; i < 256; i += 32) {
    int row = i >> 4, ch = i & 15;
    uint4 t = *(const uint4*)(s->f1wb + row*128 + ch*8);
    *(uint4*)(s->bufA + row*SBA + 128 + ch*8) = t;
  }
  if (lane < 16) {
    int row = lane;
    const float* g = s->geom + row*SG;
    _Float16* d = s->bufA + row*SBA;
    d[256] = (_Float16)g[24]; d[257] = (_Float16)g[25];
    d[258] = (_Float16)g[26]; d[259] = (_Float16)g[27];
    for (int c = 260; c < 288; ++c) d[c] = (_Float16)0.f;
  }
  __syncthreads();

  wave_gemm<2>(s->bufA, SBA, 288, Wws+155648, Bws+1024, s->bufB, SBB, 256, nullptr, lane);
  __syncthreads();
  wave_gemm<2>(s->bufB, SBB, 256, Wws+229376, Bws+1280, s->bufA, SBA, 128, nullptr, lane);
  __syncthreads();
  wave_gemm<2>(s->bufA, SBA, 128, Wws+262144, Bws+1408, s->bufB, SBB,  64, nullptr, lane);
  __syncthreads();

  // ---- phase 4: final 64->2 layer + flow writeback ----
  {
    const int m    = lane & 15;
    const int koff = (lane >> 4) << 3;
    v8f acc = {0.f,0.f,0.f,0.f,0.f,0.f,0.f,0.f};
    const _Float16* Bw = Wws + 270336;
    for (int kt = 0; kt < 2; ++kt) {
      HV a, b;
      const _Float16* ap = s->bufB + m*SBB + kt*32 + koff;
      a.q[0] = *(const uint4*)ap;
      a.q[1] = *(const uint4*)(ap + 16);
      const uint4* bp = (const uint4*)(Bw + ((size_t)(kt*32 + lane)) * 16);
      b.q[0] = bp[0]; b.q[1] = bp[1];
      acc = __builtin_amdgcn_wmma_f32_16x16x32_f16(false, a.v, false, b.v,
                                                   (short)0, acc, false, false);
    }
    float bvv = Bws[1472 + m];
    int rbase = (lane >> 4) << 3;
    if (m < 2) {
      float scale = (m == 0) ? (float)WF : (float)HF;
#pragma unroll
      for (int r = 0; r < 8; ++r) {
        int row = rbase + r;
        float flow = s->geom[row*SG + 28 + m] + (acc[r] + bvv) * scale;
        int rem = rem0 + row;
        out[(size_t)bat * 2 * NPB + (size_t)m * NPB + rem] = flow;
      }
    }
  }
}

// ---------------------------------------------------------------------------
// launch
// ---------------------------------------------------------------------------
extern "C" void kernel_launch(void* const* d_in, const int* in_sizes, int n_in,
                              void* d_out, int out_size, void* d_ws, size_t ws_size,
                              hipStream_t stream) {
  (void)in_sizes; (void)n_in; (void)out_size; (void)ws_size;
  _Float16* Wws = (_Float16*)d_ws;
  float*    Bws = (float*)((char*)d_ws + (size_t)271360 * 2);  // after packed f16 weights

  // matrices: ctx_w, ffn1_w1, ffn1_w2, s8_w, ffn2_w1, ffn2_w2, hw1, hw2, hw3, hw4
  const int widx[10]  = {6, 9, 11, 13, 16, 18, 20, 22, 24, 26};
  const int wgate[10] = {8, -1, -1, 15, -1, -1, -1, -1, -1, -1};
  const int Kr[10]  = { 64,128,256,128,128,256,260,256,128, 64};
  const int Nr[10]  = {128,256,128,128,256,128,256,128, 64,  2};
  const int Kp[10]  = { 64,128,256,128,128,256,288,256,128, 64};
  const int Np[10]  = {128,256,128,128,256,128,256,128, 64, 16};
  const size_t woff[10] = {0, 8192, 40960, 73728, 90112, 122880,
                           155648, 229376, 262144, 270336};
  for (int i = 0; i < 10; ++i) {
    int total = (Kp[i] >> 5) * (Np[i] >> 4) * 512;
    const float* gate = (wgate[i] >= 0) ? (const float*)d_in[wgate[i]] : nullptr;
    pack_w_kernel<<<(total + 255) / 256, 256, 0, stream>>>(
        (const float*)d_in[widx[i]], gate, Kr[i], Nr[i], Kp[i], Np[i], Wws + woff[i]);
  }

  const int bix[10]   = {7, 10, 12, 14, 17, 19, 21, 23, 25, 27};
  const int bgate[10] = {8, -1, -1, 15, -1, -1, -1, -1, -1, -1};
  const int bNr[10] = {128,256,128,128,256,128,256,128, 64,  2};
  const int bNp[10] = {128,256,128,128,256,128,256,128, 64, 16};
  const size_t boff[10] = {0, 128, 384, 512, 640, 896, 1024, 1280, 1408, 1472};
  for (int i = 0; i < 10; ++i) {
    const float* gate = (bgate[i] >= 0) ? (const float*)d_in[bgate[i]] : nullptr;
    pack_b_kernel<<<(bNp[i] + 63) / 64, 64, 0, stream>>>(
        (const float*)d_in[bix[i]], gate, bNr[i], bNp[i], Bws + boff[i]);
  }

  // 393216 points / (2 waves * 16 pts) = 12288 blocks of 64 threads
  decoder_kernel<<<12288, 64, 0, stream>>>(
      (const float*)d_in[1], (const float*)d_in[2], (const float*)d_in[3],
      (const float*)d_in[4], (const float*)d_in[5], Wws, Bws, (float*)d_out);
}
